// lossfunc_72026601554432
// MI455X (gfx1250) — compile-verified, matrix operand order checked by reading, not run on hardware
//
#include <hip/hip_runtime.h>
#include <hip/hip_bf16.h>

typedef __attribute__((ext_vector_type(16))) _Float16 v16h;
typedef __attribute__((ext_vector_type(8)))  _Float16 v8h;
typedef __attribute__((ext_vector_type(4)))  _Float16 v4h;
typedef __attribute__((ext_vector_type(8)))  float    v8f;

#define BATCH 4096
#define NROWS 8192          // 2*BATCH
#define DIM   128
#define INV_T 10.0f         // 1/0.1

__device__ __forceinline__ v16h combine16(v8h lo, v8h hi) {
    v16h r;
#pragma unroll
    for (int i = 0; i < 8; ++i) { r[i] = lo[i]; r[i + 8] = hi[i]; }
    return r;
}

// ---------------------------------------------------------------------------
// Kernel 0: zero the rowsum array + pos accumulator
// ---------------------------------------------------------------------------
__global__ void k_zero(float* __restrict__ rowsum, float* __restrict__ posAcc) {
    int i = blockIdx.x * 256 + threadIdx.x;
    if (i < NROWS) rowsum[i] = 0.0f;
    if (i == 0)    posAcc[0] = 0.0f;
}

// ---------------------------------------------------------------------------
// Kernel 1: L2-normalize each row of concat(out1,out2), store as fp16.
// One wave (32 lanes) per row, 4 floats per lane.
// ---------------------------------------------------------------------------
__global__ void k_normalize(const float* __restrict__ out1,
                            const float* __restrict__ out2,
                            _Float16* __restrict__ xn) {
    int wave = threadIdx.x >> 5;
    int lane = threadIdx.x & 31;
    int row  = blockIdx.x * 8 + wave;            // 1024 blocks * 8 waves = 8192

    const float* src = (row < BATCH) ? (out1 + (size_t)row * DIM)
                                     : (out2 + (size_t)(row - BATCH) * DIM);
    float4 v = *(const float4*)(src + lane * 4);
    float ss = v.x * v.x + v.y * v.y + v.z * v.z + v.w * v.w;
#pragma unroll
    for (int off = 1; off < 32; off <<= 1) ss += __shfl_xor(ss, off, 32);
    float inv = rsqrtf(ss);

    v4h h;
    h[0] = (_Float16)(v.x * inv);
    h[1] = (_Float16)(v.y * inv);
    h[2] = (_Float16)(v.z * inv);
    h[3] = (_Float16)(v.w * inv);
    *(v4h*)(xn + (size_t)row * DIM + lane * 4) = h;
}

// ---------------------------------------------------------------------------
// Kernel 2: fused symmetric self-GEMM + exp row-sum.
// Grid (64,64); lower-triangle blocks (by > bx) exit immediately — symmetry:
// an off-diagonal tile's exp values feed BOTH rowsum[rows] and rowsum[cols].
// 256 threads = 8 waves per block. Block tile: 128(M) x 128(N).
// B-tile staged in LDS and shared by all 8 waves; each wave computes 16 rows
// x 128 cols as 8 N-subtiles, K=128 via 4 x WMMA-K32.
// ---------------------------------------------------------------------------
__global__ void k_gemm_expsum(const _Float16* __restrict__ xn,
                              float* __restrict__ rowsum) {
    if (blockIdx.y > blockIdx.x) return;          // upper triangle only
    const bool diag = (blockIdx.y == blockIdx.x);

    __shared__ __align__(16) _Float16 Bs[128 * DIM];   // 32 KB
    __shared__ float colAcc[128];

    const int rowBase = blockIdx.y * 128;
    const int colBase = blockIdx.x * 128;

    // ---- cooperative stage of the 128-row B tile into LDS ----
    {
        int t    = threadIdx.x;          // 0..255
        int r    = t >> 1;               // 0..127
        int half = t & 1;                // 64 halves each
        const v8h* src = (const v8h*)(xn + (size_t)(colBase + r) * DIM + half * 64);
        v8h*       dst = (v8h*)(Bs + r * DIM + half * 64);
#pragma unroll
        for (int i = 0; i < 8; ++i) dst[i] = src[i];
    }
    if (threadIdx.x < 128) colAcc[threadIdx.x] = 0.0f;
    __syncthreads();

    const int wave = threadIdx.x >> 5;   // 0..7
    const int lane = threadIdx.x & 31;
    const int m    = lane & 15;          // N-index / A-row sub-index
    const int hi   = lane >> 4;          // K-half selector

    // ---- preload the wave's A fragments for all 4 K-chunks ----
    const int rowG = rowBase + wave * 16 + m;
    const _Float16* arow = xn + (size_t)rowG * DIM;
    v16h a[4];
#pragma unroll
    for (int kk = 0; kk < 4; ++kk) {
        v8h lo = *(const v8h*)(arow + kk * 32 + hi * 8);
        v8h hc = *(const v8h*)(arow + kk * 32 + hi * 8 + 16);
        a[kk] = combine16(lo, hc);
    }

    // per-lane row partials: index v -> row (wave*16 + v + 8*hi)
    // per-lane col partials: index n -> col (colBase + n*16 + m)
    float rsum[8];
    float csum[8];
#pragma unroll
    for (int v = 0; v < 8; ++v) { rsum[v] = 0.0f; csum[v] = 0.0f; }

#pragma unroll
    for (int n = 0; n < 8; ++n) {
        v8f acc = {};
        const _Float16* brow = Bs + (n * 16 + m) * DIM;   // column colBase+n*16+m
#pragma unroll
        for (int kk = 0; kk < 4; ++kk) {
            v8h lo = *(const v8h*)(brow + kk * 32 + hi * 8);
            v8h hc = *(const v8h*)(brow + kk * 32 + hi * 8 + 16);
            v16h b = combine16(lo, hc);
            acc = __builtin_amdgcn_wmma_f32_16x16x32_f16(
                false, a[kk], false, b, (short)0, acc, false, false);
        }
        // D layout: lane holds N = m; element v holds M = v + 8*hi
        const int colG = colBase + n * 16 + m;
        float cs = 0.0f;
#pragma unroll
        for (int v = 0; v < 8; ++v) {
            int gr  = rowBase + wave * 16 + v + 8 * hi;
            float s = acc[v] * INV_T;
            float e = (gr == colG) ? 0.0f : __expf(s);   // diagonal masked -> 0
            rsum[v] += e;
            cs      += e;
        }
        csum[n] = cs;
    }

    // ---- row partials: reduce over the 16 lanes sharing the same M set ----
#pragma unroll
    for (int v = 0; v < 8; ++v) {
        float r = rsum[v];
#pragma unroll
        for (int off = 1; off < 16; off <<= 1) r += __shfl_xor(r, off, 32);
        if (m == 0) {
            int gr = rowBase + wave * 16 + v + 8 * hi;
            atomicAdd(&rowsum[gr], r);
        }
    }

    // ---- column partials (transpose contribution), off-diagonal tiles only --
    if (!diag) {
#pragma unroll
        for (int n = 0; n < 8; ++n) {
            float c = csum[n] + __shfl_xor(csum[n], 16, 32);  // merge hi halves
            if (hi == 0) atomicAdd(&colAcc[n * 16 + m], c);   // ds_add_f32
        }
        __syncthreads();
        if (threadIdx.x < 128)
            atomicAdd(&rowsum[colBase + threadIdx.x], colAcc[threadIdx.x]);
    }
}

// ---------------------------------------------------------------------------
// Kernel 3: positive-pair dots  Σ_k dot(X[k], X[k+B])   (fp32 accumulate)
// One wave per k.
// ---------------------------------------------------------------------------
__global__ void k_posdot(const _Float16* __restrict__ xn,
                         float* __restrict__ posAcc) {
    int wave = threadIdx.x >> 5;
    int lane = threadIdx.x & 31;
    int k    = blockIdx.x * 8 + wave;            // 512 blocks * 8 = 4096

    const _Float16* ap = xn + (size_t)k * DIM + lane * 4;
    const _Float16* bp = xn + (size_t)(k + BATCH) * DIM + lane * 4;
    float s = 0.0f;
#pragma unroll
    for (int i = 0; i < 4; ++i) s += (float)ap[i] * (float)bp[i];
#pragma unroll
    for (int off = 1; off < 32; off <<= 1) s += __shfl_xor(s, off, 32);
    if (lane == 0) atomicAdd(posAcc, s);
}

// ---------------------------------------------------------------------------
// Kernel 4: finalize.  loss = ( Σ_i log(rowsum_i) - 2*INV_T*Σ dot ) * B/2
// ---------------------------------------------------------------------------
__global__ void k_finalize(const float* __restrict__ rowsum,
                           const float* __restrict__ posAcc,
                           float* __restrict__ out) {
    __shared__ float sbuf[256];
    float s = 0.0f;
    for (int i = threadIdx.x; i < NROWS; i += 256) s += __logf(rowsum[i]);
    sbuf[threadIdx.x] = s;
    __syncthreads();
    for (int w = 128; w > 0; w >>= 1) {
        if (threadIdx.x < w) sbuf[threadIdx.x] += sbuf[threadIdx.x + w];
        __syncthreads();
    }
    if (threadIdx.x == 0) {
        float lossSum = sbuf[0] - 2.0f * INV_T * posAcc[0];
        out[0] = lossSum * (0.5f * (float)BATCH);
    }
}

// ---------------------------------------------------------------------------
extern "C" void kernel_launch(void* const* d_in, const int* in_sizes, int n_in,
                              void* d_out, int out_size, void* d_ws, size_t ws_size,
                              hipStream_t stream) {
    const float* out1 = (const float*)d_in[0];
    const float* out2 = (const float*)d_in[1];
    float* out = (float*)d_out;

    // workspace layout
    char* ws = (char*)d_ws;
    _Float16* xn     = (_Float16*)ws;                                  // 2 MB
    float*    rowsum = (float*)(ws + (size_t)NROWS * DIM * sizeof(_Float16));
    float*    posAcc = rowsum + NROWS;

    k_zero<<<(NROWS + 255) / 256, 256, 0, stream>>>(rowsum, posAcc);
    k_normalize<<<NROWS / 8, 256, 0, stream>>>(out1, out2, xn);
    {
        dim3 grid(NROWS / 128, NROWS / 128);   // 64 x 64 tiles, lower half no-op
        k_gemm_expsum<<<grid, 256, 0, stream>>>(xn, rowsum);
    }
    k_posdot<<<BATCH / 8, 256, 0, stream>>>(xn, posAcc);
    k_finalize<<<1, 256, 0, stream>>>(rowsum, posAcc, out);
}
